// TrainableRelationEncoder_48653389529547
// MI455X (gfx1250) — matrix operand order repaired and checked by reading, not compiled, not run on hardware
//
#include <hip/hip_runtime.h>
#include <hip/hip_bf16.h>

// Tile/type helpers for CDNA5 WMMA (wave32, 16x16x32 f16 -> f32)
typedef __attribute__((ext_vector_type(16))) _Float16 v16h;
typedef __attribute__((ext_vector_type(8)))  _Float16 v8h;
typedef __attribute__((ext_vector_type(8)))  float    v8f;

#define DIM 64
#define LDS_ST 72      // padded row stride in halves (144 B, 16B-aligned rows)
#define EPS 0.01f

union V16U { v16h v; v8h h[2]; };

// A-fragment (16-bit A 16x32 layout): lane<16 holds row M=lane, K = k0+{0..7,16..23};
// lane>=16 holds row M=lane-16.., K = k0+{8..15,24..31}. Rows are row-major in LDS.
__device__ inline v16h load_A_frag(const _Float16* sA, int lane, int mBase, int k0) {
    int row = mBase + (lane & 15);
    int kA  = k0 + ((lane >> 4) << 3);          // +8 for upper half-wave
    V16U u;
    u.h[0] = *(const v8h*)(sA + row * LDS_ST + kA);
    u.h[1] = *(const v8h*)(sA + row * LDS_ST + kA + 16);
    return u.v;
}

// B-fragment (16-bit B 32x16): lane<16 holds col N=lane, K = k0+{0..7,16..23};
// lane>=16 holds K = k0+{8..15,24..31}. LDS buffer stores B transposed (rows = N, cols = K),
// so each lane reads two runs of 8 contiguous halves, 16 apart.
__device__ inline v16h load_B_frag(const _Float16* sBT, int lane, int nBase, int k0) {
    int col = nBase + (lane & 15);
    int kB  = k0 + ((lane >> 4) << 3);          // +8 for upper half-wave
    V16U u;
    u.h[0] = *(const v8h*)(sBT + col * LDS_ST + kB);
    u.h[1] = *(const v8h*)(sBT + col * LDS_ST + kB + 16);
    return u.v;
}

__global__ __launch_bounds__(512)
void TrainableRelationEncoder_48653389529547_kernel(
    const int*   __restrict__ relation_mapping,  // (E,2)
    const float* __restrict__ relation_embs,     // (32,64,64)
    const float* __restrict__ Q,                 // (5000,64,64)
    float*       __restrict__ out,               // (E,64,64)
    int E_)
{
    __shared__ __align__(16) _Float16 sQgT[DIM * LDS_ST];  // Qg^T : A of stage1, B^T of stage2
    __shared__ __align__(16) _Float16 sRelT[DIM * LDS_ST]; // rel^T: B^T of stage1
    __shared__ __align__(16) _Float16 sM1[DIM * LDS_ST];   // M1   : A of stage2

    int e = blockIdx.x;
    if (e >= E_) return;

    int qi = relation_mapping[2 * e + 0];
    int wi = relation_mapping[2 * e + 1];
    int ri = wi >> 1;
    float sgn = (wi & 1) ? -EPS : EPS;   // W[2r]=I+eps*A, W[2r+1]=I-eps*A

    const float* Qg = Q             + (size_t)qi * (DIM * DIM);
    const float* Rm = relation_embs + (size_t)ri * (DIM * DIM);

    int tid = threadIdx.x;

    // Cooperative load: fp32 -> f16, stored transposed. Coalesced global reads.
    // Q table (82 MB) + relation table (1 MB) live in the 192 MB L2, so the
    // per-edge gathers mostly hit L2 -- as long as we keep the output stream
    // from evicting them (see NT stores below).
#pragma unroll
    for (int t = 0; t < 8; ++t) {
        int idx = tid + t * 512;             // 4096 elements / 512 threads
        int r = idx >> 6, c = idx & 63;
        sQgT[c * LDS_ST + r] = (_Float16)Qg[idx];
        float w = ((r == c) ? 1.0f : 0.0f) + sgn * Rm[idx];
        sRelT[c * LDS_ST + r] = (_Float16)w;
    }
    __syncthreads();

    int wave = tid >> 5;
    int lane = tid & 31;
    int mBase = (wave >> 2) << 4;   // 4x4 grid of 16x16 tiles over the 64x64 output
    int nBase = (wave & 3) << 4;
    int rowHi = (lane >> 4) << 3;   // C/D layout: lanes 16-31 hold rows +8
    int colC  = lane & 15;

    // -------- Stage 1: M1 = Qg^T @ rel --------
    v8f acc = {};
#pragma unroll
    for (int k0 = 0; k0 < DIM; k0 += 32) {
        v16h a = load_A_frag(sQgT, lane, mBase, k0);
        v16h b = load_B_frag(sRelT, lane, nBase, k0);
        acc = __builtin_amdgcn_wmma_f32_16x16x32_f16(
            /*neg_a=*/false, a, /*neg_b=*/false, b,
            /*c_mod=*/(short)0, acc, /*reuse_a=*/false, /*reuse_b=*/false);
    }
#pragma unroll
    for (int v = 0; v < 8; ++v)
        sM1[(mBase + v + rowHi) * LDS_ST + nBase + colC] = (_Float16)acc[v];
    __syncthreads();

    // -------- Stage 2: out = M1 @ Qg  (B^T(l,k) = Qg(k,l) = Qg^T(l,k) -> reuse sQgT) --------
    v8f acc2 = {};
#pragma unroll
    for (int k0 = 0; k0 < DIM; k0 += 32) {
        v16h a = load_A_frag(sM1, lane, mBase, k0);
        v16h b = load_B_frag(sQgT, lane, nBase, k0);
        acc2 = __builtin_amdgcn_wmma_f32_16x16x32_f16(
            false, a, false, b, (short)0, acc2, false, false);
    }

    // Output is 492 MB, written once, never re-read by this kernel: store
    // non-temporal so the stream does not evict the L2-resident Q table.
    // Each VGPR store covers two contiguous 64B row segments across the wave.
    float* op = out + (size_t)e * (DIM * DIM);
#pragma unroll
    for (int v = 0; v < 8; ++v)
        __builtin_nontemporal_store(acc2[v], op + (mBase + v + rowHi) * DIM + nBase + colC);
}

extern "C" void kernel_launch(void* const* d_in, const int* in_sizes, int n_in,
                              void* d_out, int out_size, void* d_ws, size_t ws_size,
                              hipStream_t stream) {
    // setup_inputs order:
    // 0: relation_mapping (E,2) int32
    // 1: node_mapping     (unused)
    // 2: edge_index       (unused)
    // 3: node_embs        (unused)
    // 4: relation_embs    (32,64,64) f32
    // 5: Q                (5000,64,64) f32
    const int*   relation_mapping = (const int*)d_in[0];
    const float* relation_embs    = (const float*)d_in[4];
    const float* Q                = (const float*)d_in[5];
    float*       out              = (float*)d_out;

    int E_ = in_sizes[0] / 2;
    dim3 grid((unsigned)E_), block(512);
    hipLaunchKernelGGL(TrainableRelationEncoder_48653389529547_kernel,
                       grid, block, 0, stream,
                       relation_mapping, relation_embs, Q, out, E_);
}
